// WassersteinPredictionHead_14585708937684
// MI455X (gfx1250) — matrix operand order repaired and checked by reading, not compiled, not run on hardware
//
#include <hip/hip_runtime.h>
#include <hip/hip_bf16.h>

// ---------------------------------------------------------------------------
// Wasserstein prediction head for MI455X (gfx1250, wave32, WMMA).
//
// out[i,v] = row_const[i] + col_const[v] - 2*( hm[i,:]·e1[v,:] + s1[i,:]·s2[v,:] )
//   row_const[i] = sum(hm[i]^2) + sum(hc[i])
//   col_const[v] = sum(e1[v]^2) + sum(e2[v])
//   hm = elu(x_m W^T + b), hc = elu(x_c W^T + b)
//   e2 = elu(emb_c)+1, s1 = sqrt(max(hc,eps)), s2 = sqrt(max(e2,eps))
//
// Big GEMMs: split-bf16 (hi+lo) v_wmma_f32_16x16x32_bf16, 3 products per
// GEMM per k-step => ~f32 accuracy. Embedding conversion is done ONCE per
// element (per-block LDS staging, pre-swizzled to WMMA-B lane order) instead
// of once per wave. Memory roofline: ~307 MB @ 23.3 TB/s ~= 13 us.
// ---------------------------------------------------------------------------

#define BB   256
#define DD   256
#define VV   100001
#define CLAMP_MIN 1e-24f

typedef __attribute__((ext_vector_type(16))) __bf16 v16bf;
typedef __attribute__((ext_vector_type(8)))  __bf16 v8bf;
typedef __attribute__((ext_vector_type(8)))  float  v8f;

__device__ __forceinline__ float elu_fast(float x) {
    return x > 0.0f ? x : (__expf(x) - 1.0f);
}

// ----------------------------- Phase 1 -------------------------------------
// hm/hc = elu(x W^T + b); store hm and s1=sqrt(max(hc,eps)) as bf16 hi+lo
// streams (row-major [256][256]) + row_const[256].
__global__ __launch_bounds__(256) void wph_phase1(
    const float* __restrict__ xm, const float* __restrict__ xc,
    const float* __restrict__ W,  const float* __restrict__ bias,
    unsigned short* __restrict__ hmhi_u, unsigned short* __restrict__ hmlo_u,
    unsigned short* __restrict__ s1hi_u, unsigned short* __restrict__ s1lo_u,
    float* __restrict__ row_const)
{
    __shared__ float sxm[DD];
    __shared__ float sxc[DD];
    __shared__ float red[256];

    const int i = blockIdx.x;     // output row
    const int j = threadIdx.x;    // output column

    sxm[j] = xm[i * DD + j];
    sxc[j] = xc[i * DD + j];
    __syncthreads();

    float am = bias[j];
    float ac = bias[j];
    const float* wr = W + j * DD;
#pragma unroll 4
    for (int k = 0; k < DD; ++k) {
        const float w = wr[k];
        am = fmaf(sxm[k], w, am);
        ac = fmaf(sxc[k], w, ac);
    }
    const float hm = elu_fast(am);
    const float hc = elu_fast(ac);
    const float s1 = __builtin_amdgcn_sqrtf(fmaxf(hc, CLAMP_MIN));

    __bf16* hmhi = (__bf16*)hmhi_u; __bf16* hmlo = (__bf16*)hmlo_u;
    __bf16* s1hi = (__bf16*)s1hi_u; __bf16* s1lo = (__bf16*)s1lo_u;

    const int idx = i * DD + j;
    {   // split-bf16: x ~= hi + lo
        __bf16 h = (__bf16)hm;  hmhi[idx] = h;  hmlo[idx] = (__bf16)(hm - (float)h);
        __bf16 s = (__bf16)s1;  s1hi[idx] = s;  s1lo[idx] = (__bf16)(s1 - (float)s);
    }

    red[j] = hm * hm + hc;      // row_const contribution
    __syncthreads();
#pragma unroll
    for (int s = 128; s > 0; s >>= 1) {
        if (j < s) red[j] += red[j + s];
        __syncthreads();
    }
    if (j == 0) row_const[i] = red[0];
}

// ----------------------------- Phase 3 -------------------------------------
// One block = one 16-item N tile; 16 waves cover all 256 rows (16 M tiles).
// Stage 1: cooperatively convert the tile's embeddings into LDS, already
//          swizzled into WMMA-B per-lane operand order (once per element).
// Stage 2: K loop of 8 x 32; A from global bf16 streams (L2-resident),
//          B from LDS; 6 chained WMMAs per k-step (2 GEMMs x hi/lo split).
__device__ __forceinline__ v16bf loadA16(const __bf16* __restrict__ row, int off0)
{
    // A operand in this code's contiguous k order (B staged to match): elems 0..7 = K off0..off0+7,
    // elems 8..15 = K off0+8..off0+15 (off0 = k0 + 16*half).
    v8bf c0 = *(const v8bf*)(row + off0);
    v8bf c1 = *(const v8bf*)(row + off0 + 8);
    v16bf r;
#pragma unroll
    for (int t = 0; t < 8; ++t) { r[t] = c0[t]; r[8 + t] = c1[t]; }
    return r;
}

__global__ __launch_bounds__(512) void wph_phase3(
    const float* __restrict__ emb_m, const float* __restrict__ emb_c,
    const unsigned short* __restrict__ hmhi_u, const unsigned short* __restrict__ hmlo_u,
    const unsigned short* __restrict__ s1hi_u, const unsigned short* __restrict__ s1lo_u,
    const float* __restrict__ row_const, float* __restrict__ out)
{
    // B-operand staging: [stream][chunk=k0+lane][pos 0..15] bf16, 32 KB total.
    // Stream 0: e1_hi, 1: e1_lo, 2: s2_hi, 3: s2_lo.
    __shared__ __align__(16) __bf16 ldsB[4][256][16];
    __shared__ float ldscol[16];

    const __bf16* hmhi = (const __bf16*)hmhi_u;
    const __bf16* hmlo = (const __bf16*)hmlo_u;
    const __bf16* s1hi = (const __bf16*)s1hi_u;
    const __bf16* s1lo = (const __bf16*)s1lo_u;

    const int lane = threadIdx.x & 31;
    const int wave = threadIdx.x >> 5;

    // ---------------- Stage 1: convert tile embeddings into LDS ------------
    {
        const int citem = wave;              // 16 waves <-> 16 items
        const int d0    = lane * 8;          // dims d0..d0+7 (one 8-chunk)
        const int gitem = blockIdx.x * 16 + citem;
        const int gC    = gitem < VV ? gitem : (VV - 1);
        const float* e1r = emb_m + (size_t)gC * DD;
        const float* ecr = emb_c + (size_t)gC * DD;

        // Swizzle target: element (item n, dim d) -> chunk = (d&~31) + 16*((d>>4)&1) + n
        const int chunk = (d0 & ~31) + ((d0 >> 4) & 1) * 16 + citem;
        const int pos   = d0 & 15;           // 0 or 8 -> 16B aligned

        float a[8];
        *(float4*)&a[0] = *(const float4*)(e1r + d0);
        *(float4*)&a[4] = *(const float4*)(e1r + d0 + 4);
        float colp = 0.0f;
        v8bf vh, vl;
#pragma unroll
        for (int t = 0; t < 8; ++t) {
            const float x = a[t];
            colp = fmaf(x, x, colp);
            __bf16 h = (__bf16)x;
            vh[t] = h;
            vl[t] = (__bf16)(x - (float)h);
        }
        *(v8bf*)&ldsB[0][chunk][pos] = vh;
        *(v8bf*)&ldsB[1][chunk][pos] = vl;

        *(float4*)&a[0] = *(const float4*)(ecr + d0);
        *(float4*)&a[4] = *(const float4*)(ecr + d0 + 4);
#pragma unroll
        for (int t = 0; t < 8; ++t) {
            const float e2 = elu_fast(a[t]) + 1.0f;
            colp += e2;
            const float s = __builtin_amdgcn_sqrtf(fmaxf(e2, CLAMP_MIN));
            __bf16 h = (__bf16)s;
            vh[t] = h;
            vl[t] = (__bf16)(s - (float)h);
        }
        *(v8bf*)&ldsB[2][chunk][pos] = vh;
        *(v8bf*)&ldsB[3][chunk][pos] = vl;

        // col_const[citem]: all 32 lanes of this wave hold partials.
#pragma unroll
        for (int off = 16; off > 0; off >>= 1)
            colp += __shfl_xor(colp, off, 32);
        if (lane == 0) ldscol[citem] = colp;
    }
    __syncthreads();

    // ---------------- Stage 2: WMMA K loop ---------------------------------
    const int n16  = lane & 15;
    const int half = lane >> 4;
    const int item = blockIdx.x * 16 + n16;
    const int rowBase = wave * 16;

    const __bf16* Amh = hmhi + (rowBase + n16) * DD;   // lane&15 == A row
    const __bf16* Aml = hmlo + (rowBase + n16) * DD;
    const __bf16* Ash = s1hi + (rowBase + n16) * DD;
    const __bf16* Asl = s1lo + (rowBase + n16) * DD;

    v8f accM = {};
    v8f accC = {};

#pragma unroll 2
    for (int k0 = 0; k0 < DD; k0 += 32) {
        const int offA  = k0 + 16 * half;   // A operand chunk base
        const int chunk = k0 + lane;        // B operand chunk (pre-swizzled)

        v16bf amh = loadA16(Amh, offA);
        v16bf aml = loadA16(Aml, offA);
        v16bf ash = loadA16(Ash, offA);
        v16bf asl = loadA16(Asl, offA);

        v16bf e1h, e1l, s2h, s2l;
        {
            v8bf c0, c1;
            c0 = *(const v8bf*)&ldsB[0][chunk][0];
            c1 = *(const v8bf*)&ldsB[0][chunk][8];
#pragma unroll
            for (int t = 0; t < 8; ++t) { e1h[t] = c0[t]; e1h[8 + t] = c1[t]; }
            c0 = *(const v8bf*)&ldsB[1][chunk][0];
            c1 = *(const v8bf*)&ldsB[1][chunk][8];
#pragma unroll
            for (int t = 0; t < 8; ++t) { e1l[t] = c0[t]; e1l[8 + t] = c1[t]; }
            c0 = *(const v8bf*)&ldsB[2][chunk][0];
            c1 = *(const v8bf*)&ldsB[2][chunk][8];
#pragma unroll
            for (int t = 0; t < 8; ++t) { s2h[t] = c0[t]; s2h[8 + t] = c1[t]; }
            c0 = *(const v8bf*)&ldsB[3][chunk][0];
            c1 = *(const v8bf*)&ldsB[3][chunk][8];
#pragma unroll
            for (int t = 0; t < 8; ++t) { s2l[t] = c0[t]; s2l[8 + t] = c1[t]; }
        }

        // split-bf16 products: hi*hi + hi*lo + lo*hi
        accM = __builtin_amdgcn_wmma_f32_16x16x32_bf16(false, amh, false, e1h,
                                                       (short)0, accM, false, false);
        accM = __builtin_amdgcn_wmma_f32_16x16x32_bf16(false, amh, false, e1l,
                                                       (short)0, accM, false, false);
        accM = __builtin_amdgcn_wmma_f32_16x16x32_bf16(false, aml, false, e1h,
                                                       (short)0, accM, false, false);

        accC = __builtin_amdgcn_wmma_f32_16x16x32_bf16(false, ash, false, s2h,
                                                       (short)0, accC, false, false);
        accC = __builtin_amdgcn_wmma_f32_16x16x32_bf16(false, ash, false, s2l,
                                                       (short)0, accC, false, false);
        accC = __builtin_amdgcn_wmma_f32_16x16x32_bf16(false, asl, false, s2h,
                                                       (short)0, accC, false, false);
    }

    // ---------------- Epilogue ---------------------------------------------
    if (item < VV) {
        const float colfull = ldscol[n16];
#pragma unroll
        for (int r = 0; r < 8; ++r) {
            const int rowM = rowBase + 8 * half + r;
            const float rc = row_const[rowM];
            out[(size_t)rowM * VV + item] =
                rc + colfull - 2.0f * (accM[r] + accC[r]);
        }
    }
}

// ---------------------------------------------------------------------------
extern "C" void kernel_launch(void* const* d_in, const int* in_sizes, int n_in,
                              void* d_out, int out_size, void* d_ws, size_t ws_size,
                              hipStream_t stream)
{
    const float* x_m   = (const float*)d_in[0];
    const float* x_c   = (const float*)d_in[1];
    // d_in[2] = b_seq: unused by the reference forward
    const float* W     = (const float*)d_in[3];
    const float* bias  = (const float*)d_in[4];
    const float* emb_m = (const float*)d_in[5];
    const float* emb_c = (const float*)d_in[6];
    float* out = (float*)d_out;

    // Workspace carve-up: 4 bf16 streams [256*256] + row_const[256]
    unsigned short* hmhi = (unsigned short*)d_ws;
    unsigned short* hmlo = hmhi + BB * DD;
    unsigned short* s1hi = hmlo + BB * DD;
    unsigned short* s1lo = s1hi + BB * DD;
    float* row_const     = (float*)(s1lo + BB * DD);

    wph_phase1<<<BB, 256, 0, stream>>>(x_m, x_c, W, bias,
                                       hmhi, hmlo, s1hi, s1lo, row_const);

    const int ntiles = (VV + 15) / 16;   // 6251
    wph_phase3<<<ntiles, 512, 0, stream>>>(emb_m, emb_c,
                                           hmhi, hmlo, s1hi, s1lo,
                                           row_const, out);
}
